// CPP_80676665688885
// MI455X (gfx1250) — matrix operand order repaired and checked by reading, not compiled, not run on hardware
//
#include <hip/hip_runtime.h>

// ---------------------------------------------------------------- types
typedef __bf16 bf16_t;
typedef __attribute__((ext_vector_type(16))) __bf16 bf16x16;
typedef __attribute__((ext_vector_type(8)))  __bf16 bf16x8;
typedef __attribute__((ext_vector_type(8)))  float  f32x8;

// Problem constants (from reference): B=8, C=256, CI=128, H=W=64
#define NB   8
#define NC   256
#define NCI  128
#define NSP  4096   // N = H*W
#define NM   1024   // M = (H/2)*(W/2)

// ---------------------------------------------------------------- helpers
__device__ __forceinline__ bf16_t f2bf(float f) {
    unsigned u = __builtin_bit_cast(unsigned, f);
    unsigned r = u + 0x7FFFu + ((u >> 16) & 1u);   // round-to-nearest-even
    unsigned short h = (unsigned short)(r >> 16);
    return __builtin_bit_cast(bf16_t, h);
}
__device__ __forceinline__ float bf2f(bf16_t b) {
    unsigned short h = __builtin_bit_cast(unsigned short, b);
    unsigned u = ((unsigned)h) << 16;
    return __builtin_bit_cast(float, u);
}
__device__ __forceinline__ bf16x16 cat8(bf16x8 lo, bf16x8 hi) {
    return __builtin_shufflevector(lo, hi, 0,1,2,3,4,5,6,7,8,9,10,11,12,13,14,15);
}

// A-matrix fragment (16x32 bf16, M x K).  Source is row-major [row][k], stride ld.
// ISA layout: lane L holds row M=L%16; lanes 0-15 K-groups {k0..k0+7, k0+16..k0+23},
// lanes 16-31 shifted by 8.
__device__ __forceinline__ bf16x16 load_a16(const bf16_t* base, int ld, int k0) {
    int lane = threadIdx.x & 31;
    int row  = lane & 15;
    int half = lane >> 4;
    const bf16_t* p0 = base + (size_t)row * ld + k0 + half * 8;
    bf16x8 lo = *(const bf16x8*)p0;
    bf16x8 hi = *(const bf16x8*)(p0 + 16);
    return cat8(lo, hi);
}

// B-matrix fragment (32x16 bf16, K x N).  Source gives each column n its K
// contiguously: addressed as colmajor[n][k], stride ld.  Lane L holds col N=L%16,
// lanes 0-15 K-groups {k0..k0+7, k0+16..k0+23}, lanes 16-31 shifted by 8.
__device__ __forceinline__ bf16x16 load_b16(const bf16_t* base, int ld, int k0) {
    int lane = threadIdx.x & 31;
    int col  = lane & 15;
    int half = lane >> 4;
    const bf16_t* p = base + (size_t)col * ld + k0 + half * 8;
    bf16x8 lo = *(const bf16x8*)p;
    bf16x8 hi = *(const bf16x8*)(p + 16);
    return cat8(lo, hi);
}

__device__ __forceinline__ f32x8 wmma_bf16(bf16x16 a, bf16x16 b, f32x8 c) {
    return __builtin_amdgcn_wmma_f32_16x16x32_bf16(false, a, false, b,
                                                   (short)0, c, false, false);
}

// ---------------------------------------------------------------- K0: weights -> bf16
__global__ void cvt_kernel(const float* __restrict__ Wt, const float* __restrict__ Wg,
                           const float* __restrict__ Wp, const float* __restrict__ Ww,
                           bf16_t* __restrict__ wtb, bf16_t* __restrict__ wgb,
                           bf16_t* __restrict__ wpb, bf16_t* __restrict__ wwb) {
    int i = blockIdx.x * 256 + threadIdx.x;           // 4 * 32768 elements
    if (i < 32768)            wtb[i]          = f2bf(Wt[i]);
    else if (i < 65536)       wgb[i - 32768]  = f2bf(Wg[i - 32768]);
    else if (i < 98304)       wpb[i - 65536]  = f2bf(Wp[i - 65536]);
    else if (i < 131072)      wwb[i - 98304]  = f2bf(Ww[i - 98304]);
}

// ---------------------------------------------------------------- K1: fused 3x conv1x1
// out(CI x Ntile) = W(CI x C) @ x(C x Ntile) + bias, for Wt/Wg/Wp sharing the x tile.
// theta stored (B, N, CI) row-major; conv_g/conv_p stored (B, CI, N).
__global__ __launch_bounds__(256) void proj_kernel(
    const float* __restrict__ x,
    const bf16_t* __restrict__ Wt, const bf16_t* __restrict__ Wg, const bf16_t* __restrict__ Wp,
    const float* __restrict__ bt, const float* __restrict__ bg_, const float* __restrict__ bp,
    bf16_t* __restrict__ theta, bf16_t* __restrict__ cg, bf16_t* __restrict__ cp) {
    __shared__ bf16_t xs[128][128];                   // [n][c] tile, 32 KB
    const int b    = blockIdx.y;
    const int nt   = blockIdx.x;                      // 32 tiles of 128 spatial positions
    const int tid  = threadIdx.x;
    const int wave = tid >> 5;
    const int lane = tid & 31, col = lane & 15, half = lane >> 4;
    const bf16_t* Ws[3]  = {Wt, Wg, Wp};
    const float*  bsv[3] = {bt, bg_, bp};

    #pragma unroll 1            // keep one np pass live: ~100 accum VGPRs, no MSB mode
    for (int np = 0; np < 2; ++np) {                  // 2 x 64 columns
        f32x8 acc[3][4];
        #pragma unroll
        for (int w = 0; w < 3; ++w)
            #pragma unroll
            for (int s = 0; s < 4; ++s) acc[w][s] = (f32x8){};
        #pragma unroll 1
        for (int kc = 0; kc < 2; ++kc) {              // K = 256 in 2 LDS stages
            __syncthreads();
            for (int i = tid; i < 128 * 128; i += 256) {
                int c = i >> 7, n = i & 127;
                xs[n][c] = f2bf(x[((size_t)(b * NC + kc * 128 + c)) * NSP + nt * 128 + n]);
            }
            __syncthreads();
            #pragma unroll
            for (int ks = 0; ks < 4; ++ks) {
                bf16x16 afr[3];
                #pragma unroll
                for (int w = 0; w < 3; ++w)
                    afr[w] = load_a16(Ws[w] + (size_t)(wave * 16) * NC, NC, kc * 128 + ks * 32);
                #pragma unroll
                for (int s = 0; s < 4; ++s) {
                    bf16x16 bfr = load_b16(&xs[np * 64 + s * 16][0], 128, ks * 32);
                    #pragma unroll
                    for (int w = 0; w < 3; ++w)
                        acc[w][s] = wmma_bf16(afr[w], bfr, acc[w][s]);
                }
            }
        }
        #pragma unroll
        for (int w = 0; w < 3; ++w)
            #pragma unroll
            for (int s = 0; s < 4; ++s) {
                int n = nt * 128 + np * 64 + s * 16 + col;
                #pragma unroll
                for (int v = 0; v < 8; ++v) {
                    int ci = wave * 16 + v + 8 * half;
                    float val = acc[w][s][v] + bsv[w][ci];
                    if (w == 0)
                        theta[((size_t)b * NSP + n) * NCI + ci] = f2bf(val);
                    else {
                        bf16_t* dst = (w == 1) ? cg : cp;
                        dst[((size_t)(b * NCI + ci)) * NSP + n] = f2bf(val);
                    }
                }
            }
    }
}

// ---------------------------------------------------------------- K2: 2x2 max-pool
// g -> (B, CI, M)   (columns over m contiguous: O-GEMM B operand)
// phiT -> (B, M, CI) (columns over ci contiguous: S-GEMM B operand)
__global__ void pool_kernel(const bf16_t* __restrict__ cg, const bf16_t* __restrict__ cp,
                            bf16_t* __restrict__ g, bf16_t* __restrict__ phiT) {
    int i = blockIdx.x * 256 + threadIdx.x;           // B*CI*M = 1,048,576
    if (i >= NB * NCI * NM) return;
    int m = i & (NM - 1), ci = (i >> 10) & (NCI - 1), b = i >> 17;
    int mh = m >> 5, mw = m & 31;
    size_t base = ((size_t)(b * NCI + ci)) * NSP + (size_t)(mh * 2) * 64 + mw * 2;
    float gv = fmaxf(fmaxf(bf2f(cg[base]), bf2f(cg[base + 1])),
                     fmaxf(bf2f(cg[base + 64]), bf2f(cg[base + 65])));
    float pv = fmaxf(fmaxf(bf2f(cp[base]), bf2f(cp[base + 1])),
                     fmaxf(bf2f(cp[base + 64]), bf2f(cp[base + 65])));
    g[((size_t)(b * NCI + ci)) * NM + m]   = f2bf(gv);
    phiT[((size_t)(b * NM + m)) * NCI + ci] = f2bf(pv);
}

// ---------------------------------------------------------------- K3: flash attention
// Per wave: 16 query rows.  S(16x64) = theta(16x128) @ phi^T tile, online softmax,
// O(16x128) += P(16x64) @ g(64x128).  P transposed D->A layout via per-wave LDS.
__global__ __launch_bounds__(256) void attn_kernel(
    const bf16_t* __restrict__ theta, const bf16_t* __restrict__ phiT,
    const bf16_t* __restrict__ g, bf16_t* __restrict__ y) {
    __shared__ bf16_t ldsP[8][16][64];                // 2 KB per wave
    const int b    = blockIdx.y;
    const int nt   = blockIdx.x;
    const int tid  = threadIdx.x;
    const int wave = tid >> 5;
    const int lane = tid & 31, col = lane & 15, half = lane >> 4;
    const int n0   = nt * 128 + wave * 16;

    const bf16_t* thBase = theta + ((size_t)b * NSP + n0) * NCI;
    const bf16_t* phiB   = phiT + (size_t)b * NM * NCI;
    const bf16_t* gB     = g + (size_t)b * NCI * NM;

    bf16x16 thA[4];                                   // resident query fragments
    #pragma unroll
    for (int ks = 0; ks < 4; ++ks) thA[ks] = load_a16(thBase, NCI, ks * 32);

    f32x8 O[8];
    #pragma unroll
    for (int t = 0; t < 8; ++t) O[t] = (f32x8){};
    float rmax[8], rsum[8];
    #pragma unroll
    for (int v = 0; v < 8; ++v) { rmax[v] = -3.0e38f; rsum[v] = 0.f; }

    #pragma unroll 1
    for (int mc = 0; mc < 16; ++mc) {                 // M = 1024 in chunks of 64
        if (mc < 15)
            __builtin_prefetch(phiB + (size_t)((mc + 1) * 64) * NCI, 0, 3);
        f32x8 s[4];
        #pragma unroll
        for (int sm = 0; sm < 4; ++sm) {
            f32x8 a = (f32x8){};
            #pragma unroll
            for (int ks = 0; ks < 4; ++ks)
                a = wmma_bf16(thA[ks],
                              load_b16(phiB + (size_t)(mc * 64 + sm * 16) * NCI, NCI, ks * 32),
                              a);
            s[sm] = a;
        }
        float alpha[8];
        #pragma unroll
        for (int v = 0; v < 8; ++v) {
            float lm = fmaxf(fmaxf(s[0][v], s[1][v]), fmaxf(s[2][v], s[3][v]));
            #pragma unroll
            for (int off = 8; off; off >>= 1) lm = fmaxf(lm, __shfl_xor(lm, off, 32));
            float mn = fmaxf(rmax[v], lm);
            alpha[v] = __expf(rmax[v] - mn);
            rmax[v]  = mn;
            float rs = 0.f;
            #pragma unroll
            for (int sm = 0; sm < 4; ++sm) {
                float pv = __expf(s[sm][v] - mn);
                rs += pv;
                ldsP[wave][v + 8 * half][sm * 16 + col] = f2bf(pv);
            }
            #pragma unroll
            for (int off = 8; off; off >>= 1) rs += __shfl_xor(rs, off, 32);
            rsum[v] = rsum[v] * alpha[v] + rs;
        }
        #pragma unroll
        for (int t = 0; t < 8; ++t)
            #pragma unroll
            for (int v = 0; v < 8; ++v) O[t][v] *= alpha[v];
        bf16x16 pA[2];                                // P in A layout (K = m)
        #pragma unroll
        for (int k2 = 0; k2 < 2; ++k2) pA[k2] = load_a16(&ldsP[wave][0][0], 64, k2 * 32);
        #pragma unroll
        for (int t = 0; t < 8; ++t)
            #pragma unroll
            for (int k2 = 0; k2 < 2; ++k2) {
                bf16x16 bfr = load_b16(gB + (size_t)(t * 16) * NM, NM, mc * 64 + k2 * 32);
                O[t] = wmma_bf16(pA[k2], bfr, O[t]);
            }
    }
    #pragma unroll
    for (int t = 0; t < 8; ++t)
        #pragma unroll
        for (int v = 0; v < 8; ++v) {
            int n  = n0 + v + 8 * half;
            int ci = t * 16 + col;
            y[((size_t)b * NSP + n) * NCI + ci] = f2bf(O[t][v] / rsum[v]);
        }
}

// ---------------------------------------------------------------- K4: wy = Ww @ y^T + bw
__global__ __launch_bounds__(256) void wproj_kernel(
    const bf16_t* __restrict__ y, const bf16_t* __restrict__ Ww,
    const float* __restrict__ bw, float* __restrict__ wy) {
    const int b    = blockIdx.y;
    const int nt   = blockIdx.x;
    const int tid  = threadIdx.x;
    const int wave = tid >> 5;
    const int lane = tid & 31, col = lane & 15, half = lane >> 4;

    f32x8 acc[2][8];
    #pragma unroll
    for (int cgi = 0; cgi < 2; ++cgi)
        #pragma unroll
        for (int s = 0; s < 8; ++s) acc[cgi][s] = (f32x8){};
    #pragma unroll
    for (int ks = 0; ks < 4; ++ks) {
        bf16x16 a0 = load_a16(Ww + (size_t)(wave * 16) * NCI, NCI, ks * 32);
        bf16x16 a1 = load_a16(Ww + (size_t)((wave + 8) * 16) * NCI, NCI, ks * 32);
        #pragma unroll
        for (int s = 0; s < 8; ++s) {
            bf16x16 bfr = load_b16(y + ((size_t)b * NSP + nt * 128 + s * 16) * NCI, NCI, ks * 32);
            acc[0][s] = wmma_bf16(a0, bfr, acc[0][s]);
            acc[1][s] = wmma_bf16(a1, bfr, acc[1][s]);
        }
    }
    #pragma unroll
    for (int cgi = 0; cgi < 2; ++cgi)
        #pragma unroll
        for (int s = 0; s < 8; ++s) {
            int n = nt * 128 + s * 16 + col;
            #pragma unroll
            for (int v = 0; v < 8; ++v) {
                int c = (wave + cgi * 8) * 16 + v + 8 * half;
                wy[((size_t)(b * NC + c)) * NSP + n] = acc[cgi][s][v] + bw[c];
            }
        }
}

// ---------------------------------------------------------------- K5: BN batch stats
__global__ __launch_bounds__(256) void bnstats_kernel(
    const float* __restrict__ wy, const float* __restrict__ gamma,
    const float* __restrict__ beta, float* __restrict__ scale, float* __restrict__ shift) {
    __shared__ float s1[256], s2[256];
    const int c = blockIdx.x;
    float a = 0.f, q = 0.f;
    for (int i = threadIdx.x; i < NB * NSP; i += 256) {
        int b = i >> 12, n = i & (NSP - 1);
        float v = wy[((size_t)(b * NC + c)) * NSP + n];
        a += v; q += v * v;
    }
    s1[threadIdx.x] = a; s2[threadIdx.x] = q;
    __syncthreads();
    for (int off = 128; off; off >>= 1) {
        if (threadIdx.x < off) {
            s1[threadIdx.x] += s1[threadIdx.x + off];
            s2[threadIdx.x] += s2[threadIdx.x + off];
        }
        __syncthreads();
    }
    if (threadIdx.x == 0) {
        float inv_n = 1.0f / (float)(NB * NSP);
        float mean  = s1[0] * inv_n;
        float var   = s2[0] * inv_n - mean * mean;
        float sc    = gamma[c] * rsqrtf(var + 1e-5f);
        scale[c] = sc;
        shift[c] = beta[c] - mean * sc;
    }
}

// ---------------------------------------------------------------- K6: norm + residual + global max
__global__ __launch_bounds__(256) void finalize_kernel(
    const float* __restrict__ wy, const float* __restrict__ x,
    const float* __restrict__ scale, const float* __restrict__ shift,
    float* __restrict__ out) {
    __shared__ float sm[256];
    const int bc = blockIdx.x;                        // b*256 + c
    const int c  = bc & (NC - 1);
    const float sc = scale[c], sh = shift[c];
    const size_t base = (size_t)bc * NSP;
    float m = -3.0e38f;
    for (int n = threadIdx.x; n < NSP; n += 256)
        m = fmaxf(m, wy[base + n] * sc + sh + x[base + n]);
    sm[threadIdx.x] = m;
    __syncthreads();
    for (int off = 128; off; off >>= 1) {
        if (threadIdx.x < off) sm[threadIdx.x] = fmaxf(sm[threadIdx.x], sm[threadIdx.x + off]);
        __syncthreads();
    }
    if (threadIdx.x == 0) out[bc] = sm[0];
}

// ---------------------------------------------------------------- launch
extern "C" void kernel_launch(void* const* d_in, const int* in_sizes, int n_in,
                              void* d_out, int out_size, void* d_ws, size_t ws_size,
                              hipStream_t stream) {
    const float* x     = (const float*)d_in[0];
    const float* Wg    = (const float*)d_in[1];
    const float* bg    = (const float*)d_in[2];
    const float* Wt    = (const float*)d_in[3];
    const float* bt    = (const float*)d_in[4];
    const float* Wp    = (const float*)d_in[5];
    const float* bp    = (const float*)d_in[6];
    const float* Ww    = (const float*)d_in[7];
    const float* bw    = (const float*)d_in[8];
    const float* gamma = (const float*)d_in[9];
    const float* beta  = (const float*)d_in[10];
    float* out = (float*)d_out;

    char* ws = (char*)d_ws;
    // byte offsets into scratch
    bf16_t* wt_bf = (bf16_t*)(ws + 0);          // 128x256
    bf16_t* wg_bf = (bf16_t*)(ws + 65536);      // 128x256
    bf16_t* wp_bf = (bf16_t*)(ws + 131072);     // 128x256
    bf16_t* ww_bf = (bf16_t*)(ws + 196608);     // 256x128
    bf16_t* theta = (bf16_t*)(ws + 262144);     // (B, N, CI)   8 MB
    bf16_t* cg    = (bf16_t*)(ws + 8650752);    // (B, CI, N)   8 MB
    bf16_t* cp    = (bf16_t*)(ws + 17039360);   // (B, CI, N)   8 MB
    bf16_t* gbuf  = (bf16_t*)(ws + 25427968);   // (B, CI, M)   2 MB
    bf16_t* phiT  = (bf16_t*)(ws + 27525120);   // (B, M, CI)   2 MB
    bf16_t* ybuf  = (bf16_t*)(ws + 29622272);   // (B, N, CI)   8 MB
    float*  wy    = (float*)(ws + 38010880);    // (B, C, N)   32 MB
    float*  scale = (float*)(ws + 71565312);    // C
    float*  shift = (float*)(ws + 71566336);    // C

    cvt_kernel<<<512, 256, 0, stream>>>(Wt, Wg, Wp, Ww, wt_bf, wg_bf, wp_bf, ww_bf);
    proj_kernel<<<dim3(32, NB), 256, 0, stream>>>(x, wt_bf, wg_bf, wp_bf, bt, bg, bp,
                                                  theta, cg, cp);
    pool_kernel<<<4096, 256, 0, stream>>>(cg, cp, gbuf, phiT);
    attn_kernel<<<dim3(32, NB), 256, 0, stream>>>(theta, phiT, gbuf, ybuf);
    wproj_kernel<<<dim3(32, NB), 256, 0, stream>>>(ybuf, ww_bf, bw, wy);
    bnstats_kernel<<<NC, 256, 0, stream>>>(wy, gamma, beta, scale, shift);
    finalize_kernel<<<NB * NC, 256, 0, stream>>>(wy, x, scale, shift, out);
}
